// stIntegration_24584392802320
// MI455X (gfx1250) — compile-verified, matrix operand order checked by reading, not run on hardware
//
#include <hip/hip_runtime.h>
#include <hip/hip_bf16.h>

// ---------------------------------------------------------------------------
// Problem constants (match reference)
// ---------------------------------------------------------------------------
#define NN_   30000        // nodes
#define EE_   480000       // edges
#define D0_   3000
#define D1_   256
#define D2_   30
#define BB_   2
#define KK_   20
#define NP_   (NN_ / BB_)  // 15000

typedef __attribute__((ext_vector_type(16))) __bf16 v16bf;
typedef __attribute__((ext_vector_type(8)))  float  v8f;

// round-to-nearest-even fp32 -> bf16 bit pack (storage as ushort)
__device__ __forceinline__ unsigned short f2bf(float f) {
    unsigned u = __float_as_uint(f);
    u = (u + 0x7FFFu + ((u >> 16) & 1u)) >> 16;
    return (unsigned short)u;
}

__device__ __forceinline__ float elu_f(float x) {
    return x > 0.0f ? x : (__expf(x) - 1.0f);
}

// ---------------------------------------------------------------------------
// GEMM: C(MxN) = op(A) @ B  via v_wmma_f32_16x16x32_bf16
//   A: MxK row-major fp32 (ELU_A: apply ELU while staging)
//   BT == 0 : B is KxN row-major        (C = A @ B)
//   BT == 1 : B is NxK row-major        (C = A @ B^T)
//
// Block: 256 threads (8 wave32). Block tile: 128(M) x 256(N), K step 32.
// Wave w owns rows [w*16, w*16+16); it reuses one A fragment against 16 B
// fragments (16 WMMAs per K step) -> A is streamed from HBM exactly once.
// LDS: As 128x32 bf16 (8KB) + Bs 256x32 bf16 (16KB).
// ---------------------------------------------------------------------------
template <int BT, int ELU_A>
__global__ __launch_bounds__(256)
void gemm_bf16_wmma(const float* __restrict__ A, const float* __restrict__ B,
                    float* __restrict__ C, int M, int N, int K) {
    __shared__ unsigned short As[128][32];    //  8 KB
    __shared__ unsigned short Bs[256][32];    // 16 KB (Bs[n][k])

    const int t    = threadIdx.x;
    const int wave = t >> 5;
    const int lane = t & 31;
    const int half = lane >> 4;
    const int lr   = lane & 15;

    const int m0 = blockIdx.x * 128;
    const int n0 = blockIdx.y * 256;

    v8f acc[16];
    #pragma unroll
    for (int i = 0; i < 16; ++i) acc[i] = (v8f){};

    for (int k0 = 0; k0 < K; k0 += 32) {
        // ---- stage A tile: 128x32 fp32 -> bf16 (4 float4 per thread) ----
        if ((m0 + 128 <= M) & (k0 + 32 <= K)) {
            #pragma unroll
            for (int p = 0; p < 4; ++p) {
                int q    = p * 256 + t;        // float4 slot, 8 per row
                int row  = q >> 3;
                int kloc = (q & 7) * 4;
                const float4 v = *reinterpret_cast<const float4*>(
                    A + (size_t)(m0 + row) * K + k0 + kloc);
                float x0 = v.x, x1 = v.y, x2 = v.z, x3 = v.w;
                if (ELU_A) { x0 = elu_f(x0); x1 = elu_f(x1); x2 = elu_f(x2); x3 = elu_f(x3); }
                ushort4 b4;
                b4.x = f2bf(x0); b4.y = f2bf(x1); b4.z = f2bf(x2); b4.w = f2bf(x3);
                *reinterpret_cast<ushort4*>(&As[row][kloc]) = b4;
            }
        } else {
            #pragma unroll
            for (int p = 0; p < 4; ++p) {
                int q    = p * 256 + t;
                int row  = q >> 3;
                int kloc = (q & 7) * 4;
                #pragma unroll
                for (int i = 0; i < 4; ++i) {
                    int gm = m0 + row, gk = k0 + kloc + i;
                    float x = (gm < M && gk < K) ? A[(size_t)gm * K + gk] : 0.0f;
                    if (ELU_A) x = elu_f(x);
                    As[row][kloc + i] = f2bf(x);
                }
            }
        }
        // ---- stage B tile: Bs[c][kl], 32x256 logical, 32 elems/thread ----
        #pragma unroll
        for (int p = 0; p < 32; ++p) {
            int q = p * 256 + t;               // 8192 elements
            int c, kl;
            if (BT) { c = q >> 5;  kl = q & 31; }   // coalesced along K
            else    { c = q & 255; kl = q >> 8; }   // coalesced along N
            int gk = k0 + kl, gn = n0 + c;
            float x = 0.0f;
            if (gk < K && gn < N)
                x = BT ? B[(size_t)gn * K + gk] : B[(size_t)gk * N + gn];
            Bs[c][kl] = f2bf(x);
        }
        __syncthreads();

        // ---- one A fragment, 16 B fragments, 16 WMMAs ----
        union Frag { v16bf v; unsigned short u[16]; };
        Frag fa;
        const int arow = wave * 16 + lr;
        #pragma unroll
        for (int i = 0; i < 8; ++i) {
            int kk = half * 8 + i;
            fa.u[i]     = As[arow][kk];
            fa.u[i + 8] = As[arow][16 + kk];
        }
        #pragma unroll
        for (int nt = 0; nt < 16; ++nt) {
            Frag fb;
            const int bcol = nt * 16 + lr;
            #pragma unroll
            for (int i = 0; i < 8; ++i) {
                int kk = half * 8 + i;
                fb.u[i]     = Bs[bcol][kk];
                fb.u[i + 8] = Bs[bcol][16 + kk];
            }
            acc[nt] = __builtin_amdgcn_wmma_f32_16x16x32_bf16(
                false, fa.v, false, fb.v, (short)0, acc[nt], false, false);
        }
        __syncthreads();
    }

    // ---- store C per 32-bit C/D layout: VGPR v -> M = v + half*8, N = lr ----
    const int gm0 = m0 + wave * 16 + half * 8;
    #pragma unroll
    for (int nt = 0; nt < 16; ++nt) {
        int gn = n0 + nt * 16 + lr;
        if (gn < N) {
            #pragma unroll
            for (int v = 0; v < 8; ++v) {
                int gm = gm0 + v;
                if (gm < M) C[(size_t)gm * N + gn] = acc[nt][v];
            }
        }
    }
}

// ---------------------------------------------------------------------------
// init: zero enc0/dec0, m = -inf, ssum = 0
// ---------------------------------------------------------------------------
__global__ void k_init(float* enc0, float* dec0, float* m, float* ssum, int n) {
    int i = blockIdx.x * blockDim.x + threadIdx.x;
    if (i < n) { enc0[i] = 0.0f; dec0[i] = 0.0f; }
    if (i < NN_) { ((unsigned*)m)[i] = 0xFF800000u; ssum[i] = 0.0f; }
}

// ---------------------------------------------------------------------------
// per-node attention scores: s_src[n] = h[n]·att_src, s_dst[n] = h[n]·att_dst
// ---------------------------------------------------------------------------
__global__ __launch_bounds__(256)
void k_scores(const float* __restrict__ h, const float* __restrict__ a_s,
              const float* __restrict__ a_d, float* s_src, float* s_dst) {
    __shared__ float r1[256], r2[256];
    int n = blockIdx.x, t = threadIdx.x;
    float hv = h[(size_t)n * D1_ + t];
    r1[t] = hv * a_s[t];
    r2[t] = hv * a_d[t];
    __syncthreads();
    for (int off = 128; off > 0; off >>= 1) {
        if (t < off) { r1[t] += r1[t + off]; r2[t] += r2[t + off]; }
        __syncthreads();
    }
    if (t == 0) { s_src[n] = r1[0]; s_dst[n] = r2[0]; }
}

// signed float atomic-max via int/uint ordering trick (dest init to -inf)
__device__ __forceinline__ void atomic_max_f32(float* addr, float v) {
    if (v >= 0.0f) atomicMax((int*)addr, __float_as_int(v));
    else           atomicMin((unsigned*)addr, __float_as_uint(v));
}

// edge pass A: e = leaky_relu(s_src[src] + s_dst[dst]); segment max
__global__ void k_edge_max(const long long* __restrict__ ei,
                           const float* __restrict__ s_src,
                           const float* __restrict__ s_dst,
                           float* e_buf, float* m) {
    int e = blockIdx.x * blockDim.x + threadIdx.x;
    if (e >= EE_) return;
    long long s = ei[e], d = ei[EE_ + e];
    float v = s_src[s] + s_dst[d];
    v = v > 0.0f ? v : 0.2f * v;
    e_buf[e] = v;
    atomic_max_f32(&m[d], v);
}

// edge pass B: ex = exp(e - m[dst]); segment sum
__global__ void k_edge_exp(const long long* __restrict__ ei,
                           float* e_buf, const float* __restrict__ m,
                           float* ssum) {
    int e = blockIdx.x * blockDim.x + threadIdx.x;
    if (e >= EE_) return;
    long long d = ei[EE_ + e];
    float ex = __expf(e_buf[e] - m[d]);
    e_buf[e] = ex;
    atomicAdd(&ssum[d], ex);
}

// edge pass C: alpha = ex / (sum+eps); scatter enc0 += alpha * h[src]
// one wave per edge; 32 lanes x 8 coalesced f32 atomics
__global__ __launch_bounds__(256)
void k_edge_scatter_alpha(const long long* __restrict__ ei, float* e_buf,
                          const float* __restrict__ ssum,
                          const float* __restrict__ h, float* enc0) {
    int e = blockIdx.x * 8 + (threadIdx.x >> 5);
    int lane = threadIdx.x & 31;
    if (e >= EE_) return;
    long long s = ei[e], d = ei[EE_ + e];
    float alpha = e_buf[e] / (ssum[d] + 1e-16f);
    if (lane == 0) e_buf[e] = alpha;             // keep alpha for pass D
    const float* hs = h + (size_t)s * D1_;
    float* ed = enc0 + (size_t)d * D1_;
    #pragma unroll
    for (int j = lane; j < D1_; j += 32)
        atomicAdd(&ed[j], alpha * hs[j]);
}

// edge pass D: scatter dec0 += alpha * hd[src]
__global__ __launch_bounds__(256)
void k_edge_scatter(const long long* __restrict__ ei,
                    const float* __restrict__ alpha_buf,
                    const float* __restrict__ hd, float* dec0) {
    int e = blockIdx.x * 8 + (threadIdx.x >> 5);
    int lane = threadIdx.x & 31;
    if (e >= EE_) return;
    long long s = ei[e], d = ei[EE_ + e];
    float alpha = alpha_buf[e];
    const float* hs = hd + (size_t)s * D1_;
    float* ed = dec0 + (size_t)d * D1_;
    #pragma unroll
    for (int j = lane; j < D1_; j += 32)
        atomicAdd(&ed[j], alpha * hs[j]);
}

// latent[n] = elu(enc0[n]) @ W_enc1  (256 -> 30), one wave per node
__global__ __launch_bounds__(32)
void k_latent(const float* __restrict__ enc0, const float* __restrict__ W1,
              float* __restrict__ latent) {
    __shared__ float buf[D1_];
    int n = blockIdx.x, lane = threadIdx.x;
    for (int j = lane; j < D1_; j += 32) buf[j] = elu_f(enc0[(size_t)n * D1_ + j]);
    __syncthreads();
    if (lane < D2_) {
        float acc = 0.0f;
        for (int k = 0; k < D1_; ++k) acc += buf[k] * W1[k * D2_ + lane];
        latent[(size_t)n * D2_ + lane] = acc;
    }
}

// hd[n,j] = sum_k latent[n,k] * W_enc1[j,k]   (30 -> 256), one block per node
__global__ __launch_bounds__(256)
void k_hd(const float* __restrict__ latent, const float* __restrict__ W1,
          float* __restrict__ hd) {
    __shared__ float lz[D2_];
    int n = blockIdx.x, t = threadIdx.x;
    if (t < D2_) lz[t] = latent[(size_t)n * D2_ + t];
    __syncthreads();
    float acc = 0.0f;
    #pragma unroll
    for (int k = 0; k < D2_; ++k) acc += lz[k] * W1[t * D2_ + k];
    hd[(size_t)n * D1_ + t] = acc;
}

// soft-assignment q per batch
__global__ void k_q(const float* __restrict__ latent,
                    const float* __restrict__ cent,
                    float* __restrict__ q0, float* __restrict__ q1) {
    int t = blockIdx.x * blockDim.x + threadIdx.x;
    if (t >= NN_) return;
    int b = t / NP_, p = t % NP_;
    const float* z  = latent + (size_t)t * D2_;
    const float* cb = cent + (size_t)b * KK_ * D2_;
    float q[KK_], s = 0.0f;
    #pragma unroll
    for (int k = 0; k < KK_; ++k) {
        float d2 = 0.0f;
        #pragma unroll
        for (int d = 0; d < D2_; ++d) {
            float df = z[d] - cb[k * D2_ + d];
            d2 += df * df;
        }
        float qk = 1.0f / (1.0f + d2 + 1e-6f);
        q[k] = qk; s += qk;
    }
    float inv = 1.0f / s;
    float* out = (b == 0 ? q0 : q1) + (size_t)p * KK_;
    #pragma unroll
    for (int k = 0; k < KK_; ++k) out[k] = q[k] * inv;
}

// ---------------------------------------------------------------------------
extern "C" void kernel_launch(void* const* d_in, const int* in_sizes, int n_in,
                              void* d_out, int out_size, void* d_ws, size_t ws_size,
                              hipStream_t stream) {
    const float*     features = (const float*)d_in[0];
    const long long* ei       = (const long long*)d_in[1];
    const float*     cent     = (const float*)d_in[2];
    const float*     W0       = (const float*)d_in[3];   // (3000,256)
    const float*     att_s    = (const float*)d_in[4];
    const float*     att_d    = (const float*)d_in[5];
    const float*     W1       = (const float*)d_in[6];   // (256,30)
    float* out = (float*)d_out;

    // output layout: latent | gene_recon | q0 | q1
    float* latent = out;
    float* gene   = out + (size_t)NN_ * D2_;                       //  900000
    float* q0     = gene + (size_t)NN_ * D0_;                      // +90.0M
    float* q1     = q0 + (size_t)NP_ * KK_;

    // workspace layout (floats)
    float* ws = (float*)d_ws;
    size_t o = 0;
    float* h     = ws + o; o += (size_t)NN_ * D1_;   // 7.68M floats
    float* enc0  = ws + o; o += (size_t)NN_ * D1_;
    float* hd    = ws + o; o += (size_t)NN_ * D1_;
    float* dec0  = ws + o; o += (size_t)NN_ * D1_;
    float* s_src = ws + o; o += NN_;
    float* s_dst = ws + o; o += NN_;
    float* m     = ws + o; o += NN_;
    float* ssum  = ws + o; o += NN_;
    float* e_buf = ws + o; o += EE_;

    k_init<<<(NN_ * D1_ + 255) / 256, 256, 0, stream>>>(enc0, dec0, m, ssum, NN_ * D1_);

    // h = features @ W_enc0  (NN); full N=256 per block -> features read once
    {
        dim3 grid((NN_ + 127) / 128, (D1_ + 255) / 256);
        gemm_bf16_wmma<0, 0><<<grid, 256, 0, stream>>>(features, W0, h,
                                                       NN_, D1_, D0_);
    }

    k_scores<<<NN_, 256, 0, stream>>>(h, att_s, att_d, s_src, s_dst);

    k_edge_max<<<(EE_ + 255) / 256, 256, 0, stream>>>(ei, s_src, s_dst, e_buf, m);
    k_edge_exp<<<(EE_ + 255) / 256, 256, 0, stream>>>(ei, e_buf, m, ssum);
    k_edge_scatter_alpha<<<(EE_ + 7) / 8, 256, 0, stream>>>(ei, e_buf, ssum, h, enc0);

    k_latent<<<NN_, 32, 0, stream>>>(enc0, W1, latent);
    k_hd<<<NN_, 256, 0, stream>>>(latent, W1, hd);

    k_edge_scatter<<<(EE_ + 7) / 8, 256, 0, stream>>>(ei, e_buf, hd, dec0);

    // gene_recon = elu(dec0) @ W_enc0^T (NT); dec0 is L2-resident (30.7MB)
    {
        dim3 grid((NN_ + 127) / 128, (D0_ + 255) / 256);
        gemm_bf16_wmma<1, 1><<<grid, 256, 0, stream>>>(dec0, W0, gene,
                                                       NN_, D0_, D1_);
    }

    k_q<<<(NN_ + 255) / 256, 256, 0, stream>>>(latent, cent, q0, q1);
}
